// MultiHeadAttention_62311385530522
// MI455X (gfx1250) — matrix operand order repaired and checked by reading, not compiled, not run on hardware
//
#include <hip/hip_runtime.h>
#include <hip/hip_bf16.h>
#include <math.h>

typedef _Float16 f16;
typedef _Float16 v16h __attribute__((ext_vector_type(16)));
typedef _Float16 v8h  __attribute__((ext_vector_type(8)));
typedef float    v8f  __attribute__((ext_vector_type(8)));
typedef int      v4i  __attribute__((ext_vector_type(4)));

#define D_MODEL   1024
#define NUM_HEADS 16
#define D_K       64
#define SEQ       2048
#define BATCH     2
#define MROWS     (SEQ * BATCH)   // 4096

// ---------------------------------------------------------------------------
// CDNA5 async global->LDS copy (ASYNCcnt path), with safe fallback.
// Probe round 2 established the builtin signature:
//   (int4 addrspace(1)*, int4 addrspace(3)*, imm offset, imm cpol)
// ---------------------------------------------------------------------------
#if __has_builtin(__builtin_amdgcn_global_load_async_to_lds_b128)
#define HAVE_ASYNC_LDS 1
#else
#define HAVE_ASYNC_LDS 0
#endif

static __device__ __forceinline__ void async_copy_b128(const f16* g, f16* l) {
#if HAVE_ASYNC_LDS
  __builtin_amdgcn_global_load_async_to_lds_b128(
      (__attribute__((address_space(1))) v4i*)(v4i*)(void*)g,
      (__attribute__((address_space(3))) v4i*)(v4i*)(void*)l, 0, 0);
#else
  *(v8h*)l = *(const v8h*)g;
#endif
}

static __device__ __forceinline__ void async_wait_all() {
#if HAVE_ASYNC_LDS
#if __has_builtin(__builtin_amdgcn_s_wait_asynccnt)
  __builtin_amdgcn_s_wait_asynccnt(0);
#else
  asm volatile("s_wait_asynccnt 0" ::: "memory");
#endif
#endif
}

// ---------------------------------------------------------------------------
// WMMA wrapper: D = A(16x32 f16) * B(32x16 f16) + C(16x16 f32)
// ---------------------------------------------------------------------------
static __device__ __forceinline__ v8f wmma_f16(v16h a, v16h b, v8f c) {
  return __builtin_amdgcn_wmma_f32_16x16x32_f16(
      false, a, false, b, (short)0, c, false, false);
}

// A-matrix 16x32 fragment from row-major f16 [rows x ld], tile at (row0,k0).
// Lanes 0-15: K=k0+0..7 / k0+16..23; lanes 16-31: K=k0+8..15 / k0+24..31.
static __device__ __forceinline__ v16h load_frag_a(const f16* __restrict__ src,
                                                   int ld, int row0, int k0,
                                                   int lane) {
  int r  = row0 + (lane & 15);
  int kb = k0 + ((lane >> 4) << 3);
  const f16* p = src + (size_t)r * ld + kb;
  v8h lo = *(const v8h*)(p);
  v8h hi = *(const v8h*)(p + 16);
  return __builtin_shufflevector(lo, hi, 0, 1, 2, 3, 4, 5, 6, 7,
                                 8, 9, 10, 11, 12, 13, 14, 15);
}

// B-matrix 32x16 fragment; source holds B^T row-major (row n = column n of B,
// contiguous in K). Lanes 0-15: K=k0+0..7 / k0+16..23; lanes 16-31: K=k0+8..15 / k0+24..31.
static __device__ __forceinline__ v16h load_frag_b(const f16* __restrict__ src,
                                                   int ld, int row0, int k0,
                                                   int lane) {
  int r  = row0 + (lane & 15);
  int kb = k0 + ((lane >> 4) << 3);
  const f16* p = src + (size_t)r * ld + kb;
  v8h lo = *(const v8h*)(p);
  v8h hi = *(const v8h*)(p + 16);
  return __builtin_shufflevector(lo, hi, 0, 1, 2, 3, 4, 5, 6, 7,
                                 8, 9, 10, 11, 12, 13, 14, 15);
}

// ---------------------------------------------------------------------------
// fp32 -> f16 cast prologue (trivial: ~50MB at 23.3 TB/s)
// ---------------------------------------------------------------------------
__global__ void __launch_bounds__(256) cast_f32_f16_kernel(
    const float* __restrict__ in, f16* __restrict__ out, int n) {
  for (int i = blockIdx.x * blockDim.x + threadIdx.x; i < n;
       i += gridDim.x * blockDim.x)
    out[i] = (f16)in[i];
}

// ---------------------------------------------------------------------------
// Projection GEMM: out = X @ W^T + bias, scatter to per-head layout.
// Block = 4 waves stacked on M; block tile 256(M) x 32(N).
// Whole W strip [32 x 1024] (64 KB) staged in LDS once via async copies and
// shared by all waves. Each wave: 64x32 tile, 8 WMMA per k-step.
// vmode==0: out[((b*H+h)*SEQ + s)*64 + d]   (Q, K)
// vmode==1: out[((b*H+h)*64 + d)*SEQ + s]   (V transposed)
// ---------------------------------------------------------------------------
__global__ void __launch_bounds__(128, 1) proj_kernel(
    const f16* __restrict__ X, const f16* __restrict__ W,
    const float* __restrict__ bias, f16* __restrict__ out, int vmode) {
  __shared__ f16 wlds[32 * D_MODEL];  // 64 KB
  int tid = threadIdx.x;
  int lane = tid & 31;
  int wave = tid >> 5;
  int n0 = blockIdx.x * 32;
  int m0 = blockIdx.y * 256 + wave * 64;

  // Stage W[n0..n0+31][:] -> LDS. 4096 b128 chunks over 128 threads.
  for (int i = 0; i < 32; ++i)
    async_copy_b128(W + (size_t)(n0 + i) * D_MODEL + tid * 8,
                    &wlds[i * D_MODEL + tid * 8]);
  async_wait_all();
  __syncthreads();

  v8f c[4][2] = {};
  for (int k0 = 0; k0 < D_MODEL; k0 += 32) {
    v16h b0 = load_frag_b(wlds, D_MODEL, 0, k0, lane);
    v16h b1 = load_frag_b(wlds, D_MODEL, 16, k0, lane);
    v16h a0 = load_frag_a(X, D_MODEL, m0, k0, lane);
    v16h a1 = load_frag_a(X, D_MODEL, m0 + 16, k0, lane);
    v16h a2 = load_frag_a(X, D_MODEL, m0 + 32, k0, lane);
    v16h a3 = load_frag_a(X, D_MODEL, m0 + 48, k0, lane);
    c[0][0] = wmma_f16(a0, b0, c[0][0]);
    c[0][1] = wmma_f16(a0, b1, c[0][1]);
    c[1][0] = wmma_f16(a1, b0, c[1][0]);
    c[1][1] = wmma_f16(a1, b1, c[1][1]);
    c[2][0] = wmma_f16(a2, b0, c[2][0]);
    c[2][1] = wmma_f16(a2, b1, c[2][1]);
    c[3][0] = wmma_f16(a3, b0, c[3][0]);
    c[3][1] = wmma_f16(a3, b1, c[3][1]);
  }

  int rbase = (lane >> 4) << 3;
  for (int mt = 0; mt < 4; ++mt)
    for (int nt = 0; nt < 2; ++nt) {
      int n = n0 + nt * 16 + (lane & 15);
      float bv = bias[n];
      int h = n >> 6, d = n & 63;
      for (int i = 0; i < 8; ++i) {
        int m = m0 + mt * 16 + rbase + i;
        int s = m >> 1, b = m & 1;
        float v = c[mt][nt][i] + bv;
        if (vmode)
          out[(((size_t)(b * NUM_HEADS + h) * D_K + d) * SEQ) + s] = (f16)v;
        else
          out[(((size_t)(b * NUM_HEADS + h) * SEQ + s) * D_K) + d] = (f16)v;
      }
    }
}

// ---------------------------------------------------------------------------
// Flash attention. Block = 4 waves, one (b,h), 128 queries (32 per wave).
// K/V 32-key tiles double-buffered in LDS via async copies, shared by the
// whole block: wait -> barrier -> prefetch-next -> compute. Per key tile per
// wave: 16 WMMA (8 for S = Q K^T over 2 query blocks, 8 for O += P V).
// ---------------------------------------------------------------------------
__global__ void __launch_bounds__(128, 1) attn_kernel(
    const f16* __restrict__ qbuf, const f16* __restrict__ kbuf,
    const f16* __restrict__ vtbuf, f16* __restrict__ ctx) {
  __shared__ f16 ktile[2][32 * D_K];   // 2 x 4 KB  [key][d]
  __shared__ f16 vtile[2][D_K * 32];   // 2 x 4 KB  [d][key]
  __shared__ f16 plds_all[4][32 * 32]; // 8 KB, per-wave P staging

  int tid = threadIdx.x;
  int lane = tid & 31;
  int wave = tid >> 5;
  f16* plds = plds_all[wave];

  int bh = blockIdx.y;
  int b = bh >> 4, h = bh & 15;
  int qbase = blockIdx.x * 128 + wave * 32;

  const f16* Q = qbuf + (size_t)bh * SEQ * D_K;
  const f16* K = kbuf + (size_t)bh * SEQ * D_K;
  const f16* V = vtbuf + (size_t)bh * D_K * SEQ;  // [64][SEQ]

  // Persistent Q fragments: 2 query blocks x 2 d-halves.
  v16h qa[2][2];
  for (int qb = 0; qb < 2; ++qb)
    for (int dh = 0; dh < 2; ++dh)
      qa[qb][dh] = load_frag_a(Q, D_K, qbase + qb * 16, dh * 32, lane);

  v8f o[2][4] = {};
  float mrow[2][8], lrow[2][8], alpha[2][8];
  for (int qb = 0; qb < 2; ++qb)
    for (int i = 0; i < 8; ++i) { mrow[qb][i] = -1e30f; lrow[qb][i] = 0.0f; }
  const float scale = 0.125f;  // 1/sqrt(64)
  int col = lane & 15;
  int rbase = (lane >> 4) << 3;

  // --- tile copy helper: K tile 32x64 (8 chunks/row), Vt tile 64x32 (4/row)
  auto copy_tiles = [&](int kk, int buf) {
    for (int i = 0; i < 2; ++i) {
      int id = tid + i * 128;
      int kr = id >> 3, kc = (id & 7) * 8;
      async_copy_b128(K + (size_t)(kk + kr) * D_K + kc,
                      &ktile[buf][kr * D_K + kc]);
      int vr = id >> 2, vc = (id & 3) * 8;
      async_copy_b128(V + (size_t)vr * SEQ + kk + vc,
                      &vtile[buf][vr * 32 + vc]);
    }
  };

  copy_tiles(0, 0);

  for (int kk = 0; kk < SEQ; kk += 32) {
    int cur = (kk >> 5) & 1;
    async_wait_all();
    __syncthreads();                      // tiles ready; prev reads done
    if (kk + 32 < SEQ) copy_tiles(kk + 32, cur ^ 1);

    // K fragments from LDS, shared across both query blocks.
    v16h kb00 = load_frag_b(ktile[cur], D_K, 0, 0, lane);
    v16h kb01 = load_frag_b(ktile[cur], D_K, 0, 32, lane);
    v16h kb10 = load_frag_b(ktile[cur], D_K, 16, 0, lane);
    v16h kb11 = load_frag_b(ktile[cur], D_K, 16, 32, lane);

    for (int qb = 0; qb < 2; ++qb) {
      v8f s0 = {}, s1 = {};
      s0 = wmma_f16(qa[qb][0], kb00, s0);
      s0 = wmma_f16(qa[qb][1], kb01, s0);
      s1 = wmma_f16(qa[qb][0], kb10, s1);
      s1 = wmma_f16(qa[qb][1], kb11, s1);

      float p0[8], p1[8], rmax[8];
      for (int i = 0; i < 8; ++i) {
        p0[i] = s0[i] * scale;
        p1[i] = s1[i] * scale;
        rmax[i] = fmaxf(p0[i], p1[i]);
      }
      for (int mask = 1; mask < 16; mask <<= 1)
        for (int i = 0; i < 8; ++i)
          rmax[i] = fmaxf(rmax[i], __shfl_xor(rmax[i], mask, 32));

      float rsum[8];
      for (int i = 0; i < 8; ++i) {
        float mn = fmaxf(mrow[qb][i], rmax[i]);
        alpha[qb][i] = __expf(mrow[qb][i] - mn);
        mrow[qb][i] = mn;
        p0[i] = __expf(p0[i] - mn);
        p1[i] = __expf(p1[i] - mn);
        rsum[i] = p0[i] + p1[i];
      }
      for (int mask = 1; mask < 16; mask <<= 1)
        for (int i = 0; i < 8; ++i)
          rsum[i] += __shfl_xor(rsum[i], mask, 32);
      for (int i = 0; i < 8; ++i)
        lrow[qb][i] = lrow[qb][i] * alpha[qb][i] + rsum[i];

      for (int i = 0; i < 8; ++i) {
        plds[(qb * 16 + rbase + i) * 32 + col] = (f16)p0[i];
        plds[(qb * 16 + rbase + i) * 32 + 16 + col] = (f16)p1[i];
      }
    }

    // Same-wave LDS RAW is in-order; belt for flat lowering.
    asm volatile("s_wait_dscnt 0" ::: "memory");
    v16h pa0 = load_frag_a(plds, 32, 0, 0, lane);
    v16h pa1 = load_frag_a(plds, 32, 16, 0, lane);

    for (int nt = 0; nt < 4; ++nt) {
      v16h vb = load_frag_b(vtile[cur], 32, nt * 16, 0, lane);
      for (int i = 0; i < 8; ++i) o[0][nt][i] *= alpha[0][i];
      o[0][nt] = wmma_f16(pa0, vb, o[0][nt]);
      for (int i = 0; i < 8; ++i) o[1][nt][i] *= alpha[1][i];
      o[1][nt] = wmma_f16(pa1, vb, o[1][nt]);
    }
  }

  // Finalize: O/l -> ctx[(s*B+b)*1024 + h*64 + d] (f16)
  for (int qb = 0; qb < 2; ++qb)
    for (int nt = 0; nt < 4; ++nt)
      for (int i = 0; i < 8; ++i) {
        float v = o[qb][nt][i] / lrow[qb][i];
        int s = qbase + qb * 16 + rbase + i;
        int n = h * 64 + nt * 16 + col;
        ctx[(size_t)(s * BATCH + b) * D_MODEL + n] = (f16)v;
      }
}

// ---------------------------------------------------------------------------
// Output GEMM: d_out = ctx @ Wo^T + bo (fp32, [S,B,D]); same blocking as proj.
// ---------------------------------------------------------------------------
__global__ void __launch_bounds__(128, 1) out_proj_kernel(
    const f16* __restrict__ X, const f16* __restrict__ W,
    const float* __restrict__ bias, float* __restrict__ out) {
  __shared__ f16 wlds[32 * D_MODEL];  // 64 KB
  int tid = threadIdx.x;
  int lane = tid & 31;
  int wave = tid >> 5;
  int n0 = blockIdx.x * 32;
  int m0 = blockIdx.y * 256 + wave * 64;

  for (int i = 0; i < 32; ++i)
    async_copy_b128(W + (size_t)(n0 + i) * D_MODEL + tid * 8,
                    &wlds[i * D_MODEL + tid * 8]);
  async_wait_all();
  __syncthreads();

  v8f c[4][2] = {};
  for (int k0 = 0; k0 < D_MODEL; k0 += 32) {
    v16h b0 = load_frag_b(wlds, D_MODEL, 0, k0, lane);
    v16h b1 = load_frag_b(wlds, D_MODEL, 16, k0, lane);
    v16h a0 = load_frag_a(X, D_MODEL, m0, k0, lane);
    v16h a1 = load_frag_a(X, D_MODEL, m0 + 16, k0, lane);
    v16h a2 = load_frag_a(X, D_MODEL, m0 + 32, k0, lane);
    v16h a3 = load_frag_a(X, D_MODEL, m0 + 48, k0, lane);
    c[0][0] = wmma_f16(a0, b0, c[0][0]);
    c[0][1] = wmma_f16(a0, b1, c[0][1]);
    c[1][0] = wmma_f16(a1, b0, c[1][0]);
    c[1][1] = wmma_f16(a1, b1, c[1][1]);
    c[2][0] = wmma_f16(a2, b0, c[2][0]);
    c[2][1] = wmma_f16(a2, b1, c[2][1]);
    c[3][0] = wmma_f16(a3, b0, c[3][0]);
    c[3][1] = wmma_f16(a3, b1, c[3][1]);
  }

  int rbase = (lane >> 4) << 3;
  for (int mt = 0; mt < 4; ++mt)
    for (int nt = 0; nt < 2; ++nt) {
      int n = n0 + nt * 16 + (lane & 15);
      float bv = bias[n];
      for (int i = 0; i < 8; ++i) {
        int m = m0 + mt * 16 + rbase + i;
        out[(size_t)m * D_MODEL + n] = c[mt][nt][i] + bv;
      }
    }
}

// ---------------------------------------------------------------------------
extern "C" void kernel_launch(void* const* d_in, const int* in_sizes, int n_in,
                              void* d_out, int out_size, void* d_ws,
                              size_t ws_size, hipStream_t stream) {
  const float* query = (const float*)d_in[0];
  const float* key   = (const float*)d_in[1];
  const float* value = (const float*)d_in[2];
  const float* Wq = (const float*)d_in[3];
  const float* bq = (const float*)d_in[4];
  const float* Wk = (const float*)d_in[5];
  const float* bk = (const float*)d_in[6];
  const float* Wv = (const float*)d_in[7];
  const float* bv = (const float*)d_in[8];
  const float* Wo = (const float*)d_in[9];
  const float* bo = (const float*)d_in[10];

  const size_t SZ_X = (size_t)MROWS * D_MODEL * sizeof(f16);    // 8 MB
  const size_t SZ_W = (size_t)D_MODEL * D_MODEL * sizeof(f16);  // 2 MB

  char* ws = (char*)d_ws;
  f16* xq  = (f16*)(ws); ws += SZ_X;
  f16* xk  = (f16*)(ws); ws += SZ_X;
  f16* xv  = (f16*)(ws); ws += SZ_X;
  f16* wqh = (f16*)(ws); ws += SZ_W;
  f16* wkh = (f16*)(ws); ws += SZ_W;
  f16* wvh = (f16*)(ws); ws += SZ_W;
  f16* woh = (f16*)(ws); ws += SZ_W;
  f16* qbuf = (f16*)(ws); ws += SZ_X;
  f16* kbuf = (f16*)(ws); ws += SZ_X;
  f16* vtbuf = (f16*)(ws); ws += SZ_X;
  f16* ctx = (f16*)(ws); ws += SZ_X;  // total 64 MB

  const int nX = MROWS * D_MODEL;
  const int nW = D_MODEL * D_MODEL;
  cast_f32_f16_kernel<<<1024, 256, 0, stream>>>(query, xq, nX);
  cast_f32_f16_kernel<<<1024, 256, 0, stream>>>(key, xk, nX);
  cast_f32_f16_kernel<<<1024, 256, 0, stream>>>(value, xv, nX);
  cast_f32_f16_kernel<<<1024, 256, 0, stream>>>(Wq, wqh, nW);
  cast_f32_f16_kernel<<<1024, 256, 0, stream>>>(Wk, wkh, nW);
  cast_f32_f16_kernel<<<1024, 256, 0, stream>>>(Wv, wvh, nW);
  cast_f32_f16_kernel<<<1024, 256, 0, stream>>>(Wo, woh, nW);

  dim3 gproj(D_MODEL / 32, MROWS / 256);  // (32, 16)
  proj_kernel<<<gproj, 128, 0, stream>>>(xq, wqh, bq, qbuf, 0);
  proj_kernel<<<gproj, 128, 0, stream>>>(xk, wkh, bk, kbuf, 0);
  proj_kernel<<<gproj, 128, 0, stream>>>(xv, wvh, bv, vtbuf, 1);

  dim3 gattn(SEQ / 128, BATCH * NUM_HEADS);  // (16, 32)
  attn_kernel<<<gattn, 128, 0, stream>>>(qbuf, kbuf, vtbuf, ctx);

  out_proj_kernel<<<gproj, 128, 0, stream>>>(ctx, woh, bo, (float*)d_out);
}
